// LastMeanPositionMessageAggregator_58351425683901
// MI455X (gfx1250) — compile-verified, matrix operand order checked by reading, not run on hardware
//
#include <hip/hip_runtime.h>
#include <hip/hip_bf16.h>

// ---- problem constants (from the reference) -------------------------------
#define POS_DIM   128           // POSITION_DIM
#define TPB       256           // threads per block (8 wave32 waves)

typedef __attribute__((ext_vector_type(16))) _Float16 v16h;
typedef __attribute__((ext_vector_type(8)))  float    v8f;

// Exact per-lane identity through the matrix pipe: D = A*B + C with A=B=0.
// D's VGPR layout == C's VGPR layout, so d[i] == c[i] for every lane.
// Must be executed with EXEC all-ones within the wave (caller guarantees a
// wave-uniform branch).
__device__ __forceinline__ float wmma_identity_f32(float x) {
    v16h za = {};            // zero A (16x32 f16)
    v16h zb = {};            // zero B (32x16 f16)
    v8f  c  = {};
    c[0] = x;
    v8f d = __builtin_amdgcn_wmma_f32_16x16x32_f16(
        /*neg_a=*/false, za, /*neg_b=*/false, zb,
        /*c_mod=*/(short)0, c, /*reuse_a=*/false, /*reuse_b=*/false);
    return d[0];
}

// ---------------------------------------------------------------------------
// Kernel 1: per-call re-init (graph-replay safe).
//   - zero the pos-sum columns of out (atomic accumulation target, L2-resident)
//   - last_idx = -1, counts = 0
// ---------------------------------------------------------------------------
__global__ void lmpma_init_kernel(int* __restrict__ last_idx,
                                  int* __restrict__ counts,
                                  float* __restrict__ out,
                                  int N, int D) {
    const int OUTD = D + POS_DIM + 1;
    long long gid   = (long long)blockIdx.x * blockDim.x + threadIdx.x;
    long long total = (long long)N * POS_DIM;
    if (gid >= total) return;
    const int node = (int)(gid >> 7);      // /128
    const int p    = (int)(gid & (POS_DIM - 1));
    out[(size_t)node * OUTD + D + p] = 0.0f;
    if (p == 0) {
        last_idx[node] = -1;
        counts[node]   = 0;
    }
}

// ---------------------------------------------------------------------------
// Kernel 2: event scatter. 128 lanes per event.
//   pos[p] = m[e, D-257+p] + m[e, D-129+p]  -> atomic add into out pos cols.
//   lane p==0: atomicMax(last_idx, e), counts++.
// The 25.6MB accumulation region lives in the 192MB L2, so the float atomics
// are serviced by the L2 atomic units without HBM round-trips.
// ---------------------------------------------------------------------------
__global__ void lmpma_scatter_kernel(const int* __restrict__ node_ids,
                                     const float* __restrict__ messages,
                                     int* __restrict__ last_idx,
                                     int* __restrict__ counts,
                                     float* __restrict__ out,
                                     int B, int D) {
    const int OUTD = D + POS_DIM + 1;
    long long gid   = (long long)blockIdx.x * blockDim.x + threadIdx.x;
    long long total = (long long)B * POS_DIM;
    if (gid >= total) return;
    const int e = (int)(gid >> 7);              // event index
    const int p = (int)(gid & (POS_DIM - 1));   // position dim

    const int node = node_ids[e];
    const float* row = messages + (size_t)e * D;
    // slices: [D-257, D-129) and [D-129, D-1)
    const float a = row[D - (2 * POS_DIM + 1) + p];
    const float b = row[D - (POS_DIM + 1) + p];

    atomicAdd(&out[(size_t)node * OUTD + D + p], a + b);

    if (p == 0) {
        atomicMax(&last_idx[node], e);
        atomicAdd(&counts[node], 1);
    }
}

// ---------------------------------------------------------------------------
// Kernel 3: finalize. One 256-thread block per node.
//   - float4 (global_load/store_b128) copy of the last message row
//   - pos_mean = pos_sum / max(count,1), routed through a WMMA identity on
//     waves 0..3 (tid<128 is wave-uniform => EXEC all-ones for the wmma)
//   - last timestamp; empty nodes emit zeros everywhere.
// ---------------------------------------------------------------------------
__global__ void lmpma_finalize_kernel(const float* __restrict__ messages,
                                      const float* __restrict__ timestamps,
                                      const int* __restrict__ last_idx,
                                      const int* __restrict__ counts,
                                      float* __restrict__ out,
                                      int N, int D) {
    const int node = blockIdx.x;
    if (node >= N) return;
    const int OUTD = D + POS_DIM + 1;
    const int tid  = threadIdx.x;

    const int  last = last_idx[node];
    const bool has  = last >= 0;
    const int  safe = has ? last : 0;

    const float4* __restrict__ src = (const float4*)(messages + (size_t)safe * D);
    float4* __restrict__ dst       = (float4*)(out + (size_t)node * OUTD);

    // gfx1250 prefetch of the gathered row (global_prefetch_b8)
    __builtin_prefetch(src + tid, 0, 0);

    // last_msg copy: D/4 float4s with 256 threads (1 iter for D=1024)
    const float4 z4 = make_float4(0.f, 0.f, 0.f, 0.f);
    for (int j = tid; j < (D >> 2); j += blockDim.x) {
        float4 v = src[j];               // row 0 is valid memory even if !has
        dst[j] = has ? v : z4;
    }

    // pos_mean: waves 0..3 (full waves => EXEC all-ones inside this branch)
    if (tid < POS_DIM) {
        const size_t idx = (size_t)node * OUTD + D + tid;
        const float  s   = out[idx];                       // accumulated sum
        const float  cnt = (float)counts[node];
        float val = s / fmaxf(cnt, 1.0f);                  // 0/1 = 0 when empty
        val = wmma_identity_f32(val);                      // exact identity
        out[idx] = val;
    }

    if (tid == POS_DIM) {
        out[(size_t)node * OUTD + D + POS_DIM] = has ? timestamps[last] : 0.0f;
    }
}

// ---------------------------------------------------------------------------
extern "C" void kernel_launch(void* const* d_in, const int* in_sizes, int n_in,
                              void* d_out, int out_size, void* d_ws, size_t ws_size,
                              hipStream_t stream) {
    const int*   node_ids   = (const int*)d_in[0];
    const float* messages   = (const float*)d_in[1];
    const float* timestamps = (const float*)d_in[2];
    float*       out        = (float*)d_out;

    const int B    = in_sizes[0];            // 200000 events
    const int D    = in_sizes[1] / B;        // 1024 message dim
    const int OUTD = D + POS_DIM + 1;        // 1153
    const int N    = out_size / OUTD;        // 50000 nodes

    // workspace: last_idx[N], counts[N]  (400 KB total)
    int* last_idx = (int*)d_ws;
    int* counts   = last_idx + N;
    (void)ws_size; (void)n_in;

    {   // init
        long long total  = (long long)N * POS_DIM;
        int       blocks = (int)((total + TPB - 1) / TPB);
        lmpma_init_kernel<<<blocks, TPB, 0, stream>>>(last_idx, counts, out, N, D);
    }
    {   // scatter
        long long total  = (long long)B * POS_DIM;
        int       blocks = (int)((total + TPB - 1) / TPB);
        lmpma_scatter_kernel<<<blocks, TPB, 0, stream>>>(node_ids, messages,
                                                         last_idx, counts, out, B, D);
    }
    {   // finalize: one block per node
        lmpma_finalize_kernel<<<N, TPB, 0, stream>>>(messages, timestamps,
                                                     last_idx, counts, out, N, D);
    }
}
